// KernelNetwork_10737418240221
// MI455X (gfx1250) — matrix operand order repaired and checked by reading, not compiled, not run on hardware
//
#include <hip/hip_runtime.h>
#include <hip/hip_bf16.h>

#define PK_ROWS 256
#define PK_COLS 256
#define P_TOTAL (PK_ROWS * PK_COLS)
#define BATCH 32
#define LSTM 16
#define NEIGH 8

typedef __attribute__((ext_vector_type(16))) _Float16 v16h;
typedef __attribute__((ext_vector_type(8)))  float    v8f;

__device__ __forceinline__ float fast_rcp(float x) {
    return __builtin_amdgcn_rcpf(x);        // single v_rcp_f32 (~1 ulp)
}
__device__ __forceinline__ float sigmoidf_(float x) {
    return fast_rcp(1.0f + __expf(-x));
}
__device__ __forceinline__ float tanhf_(float x) {
    x = fminf(15.0f, fmaxf(-15.0f, x));     // med3
    const float e = __expf(2.0f * x);
    return (e - 1.0f) * fast_rcp(e + 1.0f);
}

// Combined gate weight matrix W[32][64], K-feature order:
//   K 0..15  -> Wh rows 0..15 (h), K 16 -> Wx row 0 (dyn),
//   K 17..24 -> Wx rows 1..8 (lat), K 25..31 -> 0.
__device__ __forceinline__ float gate_w(int k, int n,
                                        const float* __restrict__ Wx,
                                        const float* __restrict__ Wh) {
    if (k < 16) return Wh[k * 64 + n];
    if (k < 25) return Wx[(k - 16) * 64 + n];
    return 0.0f;
}

__global__ __launch_bounds__(256) void pk_lstm_wmma_kernel(
    const float* __restrict__ dyn_in,   // (B, P)
    const float* __restrict__ lat_in,   // (B, P, 8)
    const float* __restrict__ lat_out,  // (B, P, 8)
    const float* __restrict__ h_in,     // (B, P, 16)
    const float* __restrict__ c_in,     // (B, P, 16)
    const float* __restrict__ Wx,       // (9, 64)
    const float* __restrict__ Wh,       // (16, 64)
    const float* __restrict__ bg,       // (64)
    const float* __restrict__ Wo,       // (16, 9)
    const float* __restrict__ bo,       // (9)
    float* __restrict__ out)            // [dyn | lat | h | c] flat
{
    // Pre-swizzled f16 B tiles in WMMA register order (tiles 0..3 gate W,
    // tile 4 padded Wo). lds_B[t*32+lane] is lane's v16h B operand.
    __shared__ __attribute__((aligned(32))) v16h  lds_B[5 * 32];     // 5 KB
    __shared__ __attribute__((aligned(16))) float lds_bias[80];      // b | bo
    __shared__ __attribute__((aligned(16))) float lds_h[8][256];     // 8 KB

    // ---- stage swizzled weights once per block --------------------------
    {
        unsigned* bstage = (unsigned*)lds_B;         // 1280 packed-f16 dwords
        for (int i = threadIdx.x; i < 5 * 256; i += 256) {
            const int t  = i >> 8;          // tile
            const int ln = (i >> 3) & 31;   // lane
            const int q  = i & 7;           // dword within lane
            const int hh = ln >> 4;
            const int nn = ln & 15;
            const int k0 = hh * 8 + 2 * q + ((q >> 2) << 3); // B layout: lane col N=nn, K pair (table order)
            float w0, w1;
            if (t < 4) {
                const int n = t * 16 + nn;
                w0 = gate_w(k0,     n, Wx, Wh);
                w1 = gate_w(k0 + 1, n, Wx, Wh);
            } else {
                w0 = (k0     < 16 && nn < 9) ? Wo[k0 * 9 + nn]       : 0.0f;
                w1 = (k0 + 1 < 16 && nn < 9) ? Wo[(k0 + 1) * 9 + nn] : 0.0f;
            }
            union { _Float16 h[2]; unsigned u; } pk;
            pk.h[0] = (_Float16)w0;
            pk.h[1] = (_Float16)w1;
            bstage[i] = pk.u;
        }
        for (int i = threadIdx.x; i < 80; i += 256) {
            lds_bias[i] = (i < 64) ? bg[i] : ((i < 73) ? bo[i - 64] : 0.0f);
        }
    }
    __syncthreads();

    const int wave = threadIdx.x >> 5;
    const int lane = threadIdx.x & 31;
    const int nl   = lane & 15;   // A-row / D-column lane
    const int hf   = lane >> 4;   // K-half selector

    const int tile = blockIdx.x * 8 + wave;          // 16 rows of (B*P)
    const long long row0 = (long long)tile * 16;

    // =============== phase 1: gates GEMM ================================
    // A (16x32 f16): lane holds row M=nl.
    //   elements 0..7  -> K = hf*8 + e      -> h[hf*8 + e]
    //   elements 8..15 -> K = 16 + hf*8 + j -> hf=0: [dyn, lat0..6]
    //                                          hf=1: [lat7, 0 x7]
    const long long arow = row0 + nl;                // this lane's cell row
    const int p  = (int)(arow % P_TOTAL);
    const int gr = p >> 8;
    const int gc = p & 255;

    const float* hrow = h_in + arow * 16 + hf * 8;   // 32B aligned
    const float4 hA = *(const float4*)(hrow);
    const float4 hB = *(const float4*)(hrow + 4);

    // Neighbor offset in flattened cells is a per-direction constant:
    //   nb = arow + (dr*256 + dc). Only validity needs runtime flags.
    const bool rU = gr > 0, rD = gr < PK_ROWS - 1;
    const bool cL = gc > 0, cR = gc < PK_COLS - 1;
    const bool inbf[8] = { rU && cL, rU, rU && cR, cL,
                           cR, rD && cL, rD, rD && cR };
    const int drt[8] = {-1,-1,-1, 0, 0, 1, 1, 1};
    const int dct[8] = {-1, 0, 1,-1, 1,-1, 0, 1};
    const long long lbase = arow * 8;
    float latv[8];
    #pragma unroll
    for (int d = 0; d < 8; ++d) {
        const int doff = (drt[d] * PK_COLS + dct[d]) * 8;   // compile-time
        const float* psrc = inbf[d] ? (lat_out + doff) : lat_in;  // cndmask64
        latv[d] = psrc[lbase + d];
    }
    const float dynv = dyn_in[arow];

    v16h A;
    A[0] = (_Float16)hA.x;  A[1] = (_Float16)hA.y;
    A[2] = (_Float16)hA.z;  A[3] = (_Float16)hA.w;
    A[4] = (_Float16)hB.x;  A[5] = (_Float16)hB.y;
    A[6] = (_Float16)hB.z;  A[7] = (_Float16)hB.w;
    A[8] = (_Float16)(hf ? latv[7] : dynv);
    #pragma unroll
    for (int j = 1; j < 8; ++j)
        A[8 + j] = (_Float16)(hf ? 0.0f : latv[j - 1]);

    // 4 gate tiles, C = inline 0 (biases folded into activations below).
    const v8f zeroC = {0, 0, 0, 0, 0, 0, 0, 0};
    v8f acc[4];
    #pragma unroll
    for (int t = 0; t < 4; ++t) {
        const v16h Bt = lds_B[t * 32 + lane];
        acc[t] = __builtin_amdgcn_wmma_f32_16x16x32_f16(
            false, A, false, Bt, (short)0, zeroC, false, false);
    }

    // =============== phase 2: LSTM elementwise ==========================
    // D layout: lane nl holds gate unit nl; VGPR v -> row m = hf*8 + v.
    const long long O_LAT = (long long)BATCH * P_TOTAL;
    const long long O_H   = O_LAT + (long long)BATCH * P_TOTAL * 8;
    const long long O_C   = O_H   + (long long)BATCH * P_TOTAL * 16;

    const float bi  = lds_bias[nl];
    const float bf  = lds_bias[16 + nl];
    const float bgv = lds_bias[32 + nl];
    const float bov = lds_bias[48 + nl];

    // Hoisted per-lane bases; loops use constant instruction offsets.
    const long long eb = row0 * 16 + hf * 128 + nl;
    const float* pCin = c_in + eb;
    float*       pC   = out + O_C + eb;
    float*       pH   = out + O_H + eb;
    float*       lh   = &lds_h[wave][hf * 128 + nl];

    #pragma unroll
    for (int v = 0; v < 8; ++v) {
        const float cold = pCin[v * 16];
        const float iv = sigmoidf_(acc[0][v] + bi);
        const float fv = sigmoidf_(acc[1][v] + bf);
        const float gv = tanhf_   (acc[2][v] + bgv);
        const float ov = sigmoidf_(acc[3][v] + bov);
        const float cn = fv * cold + iv * gv;
        const float hn = ov * tanhf_(cn);
        pC[v * 16] = cn;
        pH[v * 16] = hn;
        lh[v * 16] = hn;                 // stage for D->A transpose
    }
    __syncthreads();

    // =============== phase 3: out = tanh(h_new @ Wo + bo) ===============
    const float4* hp = (const float4*)&lds_h[wave][nl * 16 + hf * 8];
    const float4 a0 = hp[0];
    const float4 a1 = hp[1];
    v16h A2;
    A2[0] = (_Float16)a0.x;  A2[1] = (_Float16)a0.y;
    A2[2] = (_Float16)a0.z;  A2[3] = (_Float16)a0.w;
    A2[4] = (_Float16)a1.x;  A2[5] = (_Float16)a1.y;
    A2[6] = (_Float16)a1.z;  A2[7] = (_Float16)a1.w;
    #pragma unroll
    for (int e = 8; e < 16; ++e) A2[e] = (_Float16)0.0f;   // K pad 16..31

    const v16h B2 = lds_B[4 * 32 + lane];
    const v8f accO = __builtin_amdgcn_wmma_f32_16x16x32_f16(
        false, A2, false, B2, (short)0, zeroC, false, false);

    const float bon = (nl < 9) ? lds_bias[64 + nl] : 0.0f;
    float* pD = out + row0 + hf * 8;                               // dyn col
    float* pL = out + O_LAT + (row0 + hf * 8) * 8 + (nl - 1);      // lat col

    #pragma unroll
    for (int v = 0; v < 8; ++v) {
        const float val = tanhf_(accO[v] + bon);
        if (nl == 0)      pD[v] = val;           // pk_dyn_out
        else if (nl <= 8) pL[v * 8] = val;       // pk_lat_out_new
    }
}

extern "C" void kernel_launch(void* const* d_in, const int* in_sizes, int n_in,
                              void* d_out, int out_size, void* d_ws, size_t ws_size,
                              hipStream_t stream) {
    (void)in_sizes; (void)n_in; (void)out_size; (void)d_ws; (void)ws_size;
    const float* dyn_in  = (const float*)d_in[0];
    const float* lat_in  = (const float*)d_in[1];
    const float* lat_out = (const float*)d_in[2];
    const float* h_in    = (const float*)d_in[3];
    const float* c_in    = (const float*)d_in[4];
    const float* Wx      = (const float*)d_in[5];
    const float* Wh      = (const float*)d_in[6];
    const float* bg      = (const float*)d_in[7];
    const float* Wo      = (const float*)d_in[8];
    const float* bo      = (const float*)d_in[9];
    // d_in[10..12] (pos0/coming_from/going_to) are recomputed analytically.
    float* out = (float*)d_out;

    const int tiles  = (BATCH * P_TOTAL) / 16;   // 131072 waves
    const int blocks = tiles / 8;                // 8 waves (256 thr) per block

    pk_lstm_wmma_kernel<<<blocks, 256, 0, stream>>>(
        dyn_in, lat_in, lat_out, h_in, c_in, Wx, Wh, bg, Wo, bo, out);
}
